// Attention_38448547234571
// MI455X (gfx1250) — matrix operand order repaired and checked by reading, not compiled, not run on hardware
//
#include <hip/hip_runtime.h>
#include <math.h>

// Problem constants (match reference)
#define BATCH 16
#define SEQ   1024     // q_len == k_len
#define EDIM  1024
#define NHEAD 8
#define HDIM  128
#define SCALE 0.08838834764831845f   // 1/sqrt(128)
#define SLOT_STRIDE ((size_t)16 * SEQ * SEQ)   // attn slot idx advances by 16 per replica

typedef __attribute__((ext_vector_type(16))) __bf16 v16bf;
typedef __attribute__((ext_vector_type(8)))  __bf16 v8bf;
typedef __attribute__((ext_vector_type(8)))  float  v8f;

union Frag16 { v16bf v; v8bf h[2]; };
union Half8  { v8bf v; __bf16 e[8]; };

static __device__ inline v8f zero8() {
  v8f z = {0.f,0.f,0.f,0.f,0.f,0.f,0.f,0.f};
  return z;
}

// ---------------------------------------------------------------------------
// K0: W_eff[e,h] = sum_n proj_w[e, n*128+h]  (fold replicated head blocks)
// ---------------------------------------------------------------------------
__global__ __launch_bounds__(256) void weff_kernel(const float* __restrict__ proj_w,
                                                   __bf16* __restrict__ weff) {
  int gid = blockIdx.x * 256 + threadIdx.x;
  if (gid >= EDIM * HDIM) return;
  int e = gid >> 7, h = gid & 127;
  float s = 0.f;
#pragma unroll
  for (int n = 0; n < NHEAD; ++n)
    s += proj_w[(size_t)e * (NHEAD * HDIM) + n * HDIM + h];
  weff[gid] = (__bf16)s;
}

// ---------------------------------------------------------------------------
// K1: Y[b] = X[b] (SEQ x EDIM, f32) @ W[b%8] (EDIM x HDIM, f32) -> bf16
// Block: 256 thr (8 waves), tile 128(M) x 128(N=HDIM), K-chunks of 32.
// ---------------------------------------------------------------------------
__global__ __launch_bounds__(256) void proj_kq_kernel(const float* __restrict__ X,
                                                      const float* __restrict__ W,
                                                      __bf16* __restrict__ Y) {
  __shared__ __align__(32) __bf16 ldsA[128 * 32];   // [m][k]
  __shared__ __align__(32) __bf16 ldsBt[128 * 32];  // [n][k] (transposed W chunk)

  const int b = blockIdx.y;
  const int hb = b & (NHEAD - 1);
  const int mBase = blockIdx.x * 128;
  const float* Xb = X + (size_t)b * SEQ * EDIM;
  const float* Wh = W + (size_t)hb * EDIM * HDIM;

  const int tid = threadIdx.x;
  const int lane = tid & 31, wv = tid >> 5;
  const int half = lane >> 4, l16 = lane & 15;

  v8f acc[8];
#pragma unroll
  for (int t = 0; t < 8; ++t) acc[t] = zero8();

  for (int kc = 0; kc < EDIM; kc += 32) {
    // stage A tile (f32 -> bf16): thread reads 16 contiguous f32 of one row
    {
      int m = tid >> 1;
      int kk0 = (tid & 1) * 16;
      const float* src = &Xb[(size_t)(mBase + m) * EDIM + kc + kk0];
      float4 f0 = *(const float4*)(src + 0);
      float4 f1 = *(const float4*)(src + 4);
      float4 f2 = *(const float4*)(src + 8);
      float4 f3 = *(const float4*)(src + 12);
      Half8 t0, t1;
      t0.e[0] = (__bf16)f0.x; t0.e[1] = (__bf16)f0.y; t0.e[2] = (__bf16)f0.z; t0.e[3] = (__bf16)f0.w;
      t0.e[4] = (__bf16)f1.x; t0.e[5] = (__bf16)f1.y; t0.e[6] = (__bf16)f1.z; t0.e[7] = (__bf16)f1.w;
      t1.e[0] = (__bf16)f2.x; t1.e[1] = (__bf16)f2.y; t1.e[2] = (__bf16)f2.z; t1.e[3] = (__bf16)f2.w;
      t1.e[4] = (__bf16)f3.x; t1.e[5] = (__bf16)f3.y; t1.e[6] = (__bf16)f3.z; t1.e[7] = (__bf16)f3.w;
      *(v8bf*)&ldsA[m * 32 + kk0 + 0] = t0.v;
      *(v8bf*)&ldsA[m * 32 + kk0 + 8] = t1.v;
      if (kc + 32 < EDIM) __builtin_prefetch(src + 32, 0, 3);
    }
    // stage W chunk transposed: thread reads 16 contiguous f32 of one W row
    {
      int kk = tid >> 3;            // 0..31
      int n0 = (tid & 7) * 16;      // 0..112
      const float* src = &Wh[(size_t)(kc + kk) * HDIM + n0];
      float vv[16];
#pragma unroll
      for (int j = 0; j < 4; ++j) {
        float4 f = *(const float4*)(src + 4 * j);
        vv[4 * j + 0] = f.x; vv[4 * j + 1] = f.y; vv[4 * j + 2] = f.z; vv[4 * j + 3] = f.w;
      }
#pragma unroll
      for (int j = 0; j < 16; ++j)
        ldsBt[(n0 + j) * 32 + kk] = (__bf16)vv[j];
    }
    __syncthreads();

    // A fragment: lane half selects K 0-15 vs 16-31 (contiguous, matches B)
    Frag16 a;
    int m = wv * 16 + l16;
    a.h[0] = *(const v8bf*)&ldsA[m * 32 + 16 * half];
    a.h[1] = *(const v8bf*)&ldsA[m * 32 + 16 * half + 8];

#pragma unroll
    for (int t = 0; t < 8; ++t) {
      Frag16 bf_;
      int n = t * 16 + l16;
      bf_.v = *(const v16bf*)&ldsBt[n * 32 + 16 * half];  // lane=N, K contiguous per half
      acc[t] = __builtin_amdgcn_wmma_f32_16x16x32_bf16(
          false, a.v, false, bf_.v, (short)0, acc[t], false, false);
    }
    __syncthreads();
  }

  __bf16* Yb = Y + (size_t)b * SEQ * HDIM;
#pragma unroll
  for (int t = 0; t < 8; ++t) {
    int col = t * 16 + l16;
#pragma unroll
    for (int r = 0; r < 8; ++r) {
      int row = mBase + wv * 16 + r + 8 * half;
      Yb[(size_t)row * HDIM + col] = (__bf16)acc[t][r];
    }
  }
}

// ---------------------------------------------------------------------------
// K2 (fused score + mask + softmax + replicate):
// Block handles a 16-row x 1024-col score band for batch b. Wave w owns cols
// [w*128,(w+1)*128). Row max/sum: shfl within 16-lane groups + LDS across the
// 8 waves. Writes normalized attn rows into all 8 replica slots (c*16+b).
// ---------------------------------------------------------------------------
__global__ __launch_bounds__(256) void score_softmax_kernel(
    const __bf16* __restrict__ qx, const __bf16* __restrict__ kx,
    const int* __restrict__ memlen, float* __restrict__ attn) {
  __shared__ float ldsMax[8 * 16];
  __shared__ float ldsSum[8 * 16];

  const int b = blockIdx.y;
  const int mBase = blockIdx.x * 16;
  const __bf16* Qb = qx + (size_t)b * SEQ * HDIM;
  const __bf16* Kb = kx + (size_t)b * SEQ * HDIM;

  const int lane = threadIdx.x & 31, wv = threadIdx.x >> 5;
  const int half = lane >> 4, l16 = lane & 15;
  const int nBase = wv * 128;

  v8f acc[8];
#pragma unroll
  for (int t = 0; t < 8; ++t) acc[t] = zero8();

#pragma unroll
  for (int kc = 0; kc < HDIM; kc += 32) {
    Frag16 a;
    int m = mBase + l16;
    a.h[0] = *(const v8bf*)&Qb[(size_t)m * HDIM + kc + 16 * half];
    a.h[1] = *(const v8bf*)&Qb[(size_t)m * HDIM + kc + 16 * half + 8];
#pragma unroll
    for (int t = 0; t < 8; ++t) {
      Frag16 bf_;
      int n = nBase + t * 16 + l16;
      bf_.v = *(const v16bf*)&Kb[(size_t)n * HDIM + kc + 16 * half];
      acc[t] = __builtin_amdgcn_wmma_f32_16x16x32_bf16(
          false, a.v, false, bf_.v, (short)0, acc[t], false, false);
    }
  }

  // scale + mask in place (lane element r -> row mBase+8*half+r, col nBase+t*16+l16)
  const int ml = memlen[b];
#pragma unroll
  for (int t = 0; t < 8; ++t) {
    int col = nBase + t * 16 + l16;
    bool valid = col < ml;
#pragma unroll
    for (int r = 0; r < 8; ++r)
      acc[t][r] = valid ? acc[t][r] * SCALE : -10000.0f;
  }

  // --- row max: per-lane over tiles, shfl over 16-lane group, LDS over waves
  float rmax[8];
#pragma unroll
  for (int r = 0; r < 8; ++r) {
    float m = acc[0][r];
#pragma unroll
    for (int t = 1; t < 8; ++t) m = fmaxf(m, acc[t][r]);
#pragma unroll
    for (int off = 8; off >= 1; off >>= 1) m = fmaxf(m, __shfl_xor(m, off, 32));
    rmax[r] = m;
  }
  if (l16 == 0) {
#pragma unroll
    for (int r = 0; r < 8; ++r) ldsMax[wv * 16 + 8 * half + r] = rmax[r];
  }
  __syncthreads();
#pragma unroll
  for (int r = 0; r < 8; ++r) {
    float m = ldsMax[8 * half + r];
#pragma unroll
    for (int w2 = 1; w2 < 8; ++w2) m = fmaxf(m, ldsMax[w2 * 16 + 8 * half + r]);
    rmax[r] = m;
  }

  // --- exp + row sum (same reduction pattern)
  float rsum[8];
#pragma unroll
  for (int r = 0; r < 8; ++r) {
    float s = 0.f;
#pragma unroll
    for (int t = 0; t < 8; ++t) { acc[t][r] = __expf(acc[t][r] - rmax[r]); s += acc[t][r]; }
#pragma unroll
    for (int off = 8; off >= 1; off >>= 1) s += __shfl_xor(s, off, 32);
    rsum[r] = s;
  }
  if (l16 == 0) {
#pragma unroll
    for (int r = 0; r < 8; ++r) ldsSum[wv * 16 + 8 * half + r] = rsum[r];
  }
  __syncthreads();
#pragma unroll
  for (int r = 0; r < 8; ++r) {
    float s = ldsSum[8 * half + r];
#pragma unroll
    for (int w2 = 1; w2 < 8; ++w2) s += ldsSum[w2 * 16 + 8 * half + r];
    rsum[r] = 1.f / s;
  }

  // --- normalize + write all 8 replica slots (idx = c*16 + b)
  float* slot0 = attn + (size_t)b * SEQ * SEQ;
#pragma unroll
  for (int t = 0; t < 8; ++t) {
    int col = nBase + t * 16 + l16;
#pragma unroll
    for (int r = 0; r < 8; ++r) {
      float v = acc[t][r] * rsum[r];
      float* p = slot0 + (size_t)(mBase + 8 * half + r) * SEQ + col;
#pragma unroll
      for (int c = 0; c < 8; ++c)
        p[c * SLOT_STRIDE] = v;
    }
  }
}

// ---------------------------------------------------------------------------
// K3: o = attn[b] @ kx[b]  (K=SEQ), then out = o @ W_eff^T + proj_b (K=HDIM)
// o (128x128) staged bf16 in LDS between the two GEMMs.
// ---------------------------------------------------------------------------
__global__ __launch_bounds__(256) void out_kernel(const float* __restrict__ attn,
                                                  const __bf16* __restrict__ kx,
                                                  const __bf16* __restrict__ weff,
                                                  const float* __restrict__ proj_b,
                                                  float* __restrict__ out) {
  __shared__ __align__(32) __bf16 ldsKt[128 * 32];   // [h][k] transposed kx chunk
  __shared__ __align__(32) __bf16 ldsO[128 * 128];   // o tile bf16

  const int b = blockIdx.y;
  const int mBase = blockIdx.x * 128;
  const float* attn_b = attn + (size_t)b * SEQ * SEQ;     // normalized slot b
  const __bf16* Kb = kx + (size_t)b * SEQ * HDIM;

  const int tid = threadIdx.x;
  const int lane = tid & 31, wv = tid >> 5;
  const int half = lane >> 4, l16 = lane & 15;

  v8f acc[8];
#pragma unroll
  for (int t = 0; t < 8; ++t) acc[t] = zero8();

  // GEMM1: o = attn @ kx
  for (int kc = 0; kc < SEQ; kc += 32) {
    // stage kx chunk transposed: thread reads 16 contiguous bf16 of one kx row
    {
      int kk = tid >> 3;           // 0..31
      int h0 = (tid & 7) * 16;     // 0..112
      Frag16 kv;
      kv.v = *(const v16bf*)&Kb[(size_t)(kc + kk) * HDIM + h0];
#pragma unroll
      for (int j = 0; j < 16; ++j)
        ldsKt[(h0 + j) * 32 + kk] = kv.v[j];
    }
    __syncthreads();

    // A fragment from f32 attn, converted to bf16 (vector f32 loads)
    Frag16 a;
    int m = mBase + wv * 16 + l16;
    const float* ap = attn_b + (size_t)m * SEQ + kc;
    if (kc + 32 < SEQ) __builtin_prefetch(ap + 32, 0, 3);
    {
      float4 f0 = *(const float4*)(ap + 16 * half);
      float4 f1 = *(const float4*)(ap + 16 * half + 4);
      float4 g0 = *(const float4*)(ap + 16 * half + 8);
      float4 g1 = *(const float4*)(ap + 16 * half + 12);
      a.v[0]  = (__bf16)f0.x; a.v[1]  = (__bf16)f0.y; a.v[2]  = (__bf16)f0.z; a.v[3]  = (__bf16)f0.w;
      a.v[4]  = (__bf16)f1.x; a.v[5]  = (__bf16)f1.y; a.v[6]  = (__bf16)f1.z; a.v[7]  = (__bf16)f1.w;
      a.v[8]  = (__bf16)g0.x; a.v[9]  = (__bf16)g0.y; a.v[10] = (__bf16)g0.z; a.v[11] = (__bf16)g0.w;
      a.v[12] = (__bf16)g1.x; a.v[13] = (__bf16)g1.y; a.v[14] = (__bf16)g1.z; a.v[15] = (__bf16)g1.w;
    }
#pragma unroll
    for (int t = 0; t < 8; ++t) {
      Frag16 bf_;
      int n = t * 16 + l16;
      bf_.v = *(const v16bf*)&ldsKt[n * 32 + 16 * half];
      acc[t] = __builtin_amdgcn_wmma_f32_16x16x32_bf16(
          false, a.v, false, bf_.v, (short)0, acc[t], false, false);
    }
    __syncthreads();
  }

  // stage o bf16
#pragma unroll
  for (int t = 0; t < 8; ++t) {
    int col = t * 16 + l16;
#pragma unroll
    for (int r = 0; r < 8; ++r)
      ldsO[(wv * 16 + r + 8 * half) * 128 + col] = (__bf16)acc[t][r];
  }
  __syncthreads();

  // GEMM2: out = o @ W_eff^T + bias  (N = EDIM, 64 n-tiles per wave row-band)
  for (int j = 0; j < 64; ++j) {
    int n0 = j * 16;
    v8f acc2 = zero8();
#pragma unroll
    for (int kb2 = 0; kb2 < 4; ++kb2) {
      int kc2 = kb2 * 32;
      Frag16 a2;
      int m = wv * 16 + l16;
      a2.h[0] = *(const v8bf*)&ldsO[m * 128 + kc2 + 16 * half];
      a2.h[1] = *(const v8bf*)&ldsO[m * 128 + kc2 + 16 * half + 8];
      Frag16 b2;
      int e = n0 + l16;
      b2.v = *(const v16bf*)&weff[(size_t)e * HDIM + kc2 + 16 * half];
      acc2 = __builtin_amdgcn_wmma_f32_16x16x32_bf16(
          false, a2.v, false, b2.v, (short)0, acc2, false, false);
    }
    int col = n0 + l16;
    float bias = proj_b[col];
#pragma unroll
    for (int r = 0; r < 8; ++r) {
      int row = mBase + wv * 16 + r + 8 * half;
      out[((size_t)b * SEQ + row) * EDIM + col] = acc2[r] + bias;
    }
  }
}

// ---------------------------------------------------------------------------
extern "C" void kernel_launch(void* const* d_in, const int* in_sizes, int n_in,
                              void* d_out, int out_size, void* d_ws, size_t ws_size,
                              hipStream_t stream) {
  const float* k      = (const float*)d_in[0];
  const float* q      = (const float*)d_in[1];
  const float* w_kx   = (const float*)d_in[2];
  const float* w_qx   = (const float*)d_in[3];
  const float* proj_w = (const float*)d_in[4];
  const float* proj_b = (const float*)d_in[5];
  const int*   memlen = (const int*)d_in[6];

  float* out  = (float*)d_out;                                  // [16,1024,1024]
  float* attn = out + (size_t)BATCH * SEQ * EDIM;               // [128,1024,1024]

  char* ws = (char*)d_ws;
  const size_t kq_bytes = (size_t)BATCH * SEQ * HDIM * sizeof(__bf16);  // 4 MB each
  __bf16* kxw  = (__bf16*)(ws);
  __bf16* qxw  = (__bf16*)(ws + kq_bytes);
  __bf16* weff = (__bf16*)(ws + 2 * kq_bytes);                  // 256 KB

  weff_kernel<<<dim3((EDIM * HDIM + 255) / 256), 256, 0, stream>>>(proj_w, weff);
  proj_kq_kernel<<<dim3(SEQ / 128, BATCH), 256, 0, stream>>>(k, w_kx, kxw);
  proj_kq_kernel<<<dim3(SEQ / 128, BATCH), 256, 0, stream>>>(q, w_qx, qxw);
  score_softmax_kernel<<<dim3(SEQ / 16, BATCH), 256, 0, stream>>>(qxw, kxw, memlen, attn);
  out_kernel<<<dim3(SEQ / 128, BATCH), 256, 0, stream>>>(attn, kxw, weff, proj_b, out);
}
